// CausalSelfAttention_43628277793399
// MI455X (gfx1250) — matrix operand order repaired and checked by reading, not compiled, not run on hardware
//
#include <hip/hip_runtime.h>
#include <hip/hip_bf16.h>
#include <math.h>

// ---------------------------------------------------------------------------
// CDNA5 (gfx1250) causal self-attention forward.
//   K1: qkv  = x @ Wqkv            (2048x1024 @ 1024x3072)  bf16 WMMA, f32 acc
//   K2: flash attention per head   (16 heads, 64q x 64k tiles, online softmax)
//   K3: out  = attn @ Wout + bias  (2048x1024 @ 1024x1024)  bf16 WMMA, f32 acc
// v_wmma_f32_16x16x32_bf16 everywhere; wave32. GEMM: 4 waves/block, 64x64 wave
// tile, ping-pong LDS + register prefetch (1 barrier per K-step). Attention:
// ping-pong K/V staging, 4-thread/row parallel online softmax.
// ---------------------------------------------------------------------------

typedef __attribute__((ext_vector_type(16))) __bf16 bf16x16;
typedef __attribute__((ext_vector_type(8)))  __bf16 bf16x8;
typedef __attribute__((ext_vector_type(4)))  __bf16 bf16x4;
typedef __attribute__((ext_vector_type(8)))  float  f32x8;

#define SEQ_LEN 2048
#define DIM     1024
#define QKV_LD  3072
#define NHEAD   16
#define HD      64

// native convert (v_cvt path on gfx1250) instead of integer-ALU RNE emulation
__device__ __forceinline__ __bf16 f2bf(float f) { return (__bf16)f; }
__device__ __forceinline__ bf16x4 f2bf4(float4 v) {
  bf16x4 r = {(__bf16)v.x, (__bf16)v.y, (__bf16)v.z, (__bf16)v.w};
  return r;
}

// A-matrix fragment (16x32 bf16): lane holds row M = lane%16.
// element e -> K = half*16 + e
__device__ __forceinline__ bf16x16 frag_a(const __bf16* p, int half) {
  bf16x8 lo = *(const bf16x8*)(p + half * 16);
  bf16x8 hi = *(const bf16x8*)(p + half * 16 + 8);
  return __builtin_shufflevector(lo, hi, 0, 1, 2, 3, 4, 5, 6, 7,
                                 8, 9, 10, 11, 12, 13, 14, 15);
}
// B-matrix fragment (32x16 bf16): lane holds col N = lane%16; e -> K = half*16+e
__device__ __forceinline__ bf16x16 frag_b(const __bf16* p, int half) {
  bf16x8 lo = *(const bf16x8*)(p + half * 16);
  bf16x8 hi = *(const bf16x8*)(p + half * 16 + 8);
  return __builtin_shufflevector(lo, hi, 0, 1, 2, 3, 4, 5, 6, 7,
                                 8, 9, 10, 11, 12, 13, 14, 15);
}

#define WMMA_BF16(a, b, c) \
  __builtin_amdgcn_wmma_f32_16x16x32_bf16(false, (a), false, (b), (short)0, (c), false, false)

// ---------------------------------------------------------------------------
// GEMM: C[M,N] = A[M,K] @ B[K,N] (+ bias). f32 memory, bf16 WMMA, f32 acc.
// Block tile 128x128, BLOCK_K=32, 128 threads = 4 waves (2x2), wave tile 64x64
// = 4x4 accumulators. Ping-pong LDS; next K-tile prefetched into registers
// before the WMMA block so global latency hides under 16 WMMAs/wave.
// ---------------------------------------------------------------------------
__global__ __launch_bounds__(128) void gemm_bf16_wmma(
    const float* __restrict__ A, const float* __restrict__ B,
    const float* __restrict__ bias, float* __restrict__ C,
    int M, int N, int K) {
  // rows padded to 40 bf16 (80B): 16B-aligned, conflict-free fragment reads
  __shared__ __align__(16) __bf16 As[2][128 * 40];  // [m][k]
  __shared__ __align__(16) __bf16 Bs[2][128 * 40];  // [n][k] (transposed stage)

  const int tid  = threadIdx.x;
  const int lane = tid & 31;
  const int wave = tid >> 5;
  const int l16  = lane & 15;
  const int half = lane >> 4;
  const int wm   = wave & 1;   // wave row (0..1) -> 64 rows
  const int wn   = wave >> 1;  // wave col (0..1) -> 64 cols
  const int m0   = blockIdx.y * 128;
  const int n0   = blockIdx.x * 128;

  // staging coords: A 128x32 (8 passes of float4), B 32x128 (8 passes)
  const int ar  = tid >> 3;         // 0..15 (+16/pass)
  const int ac4 = (tid & 7) * 4;    // 0..28
  const int bk  = tid >> 5;         // 0..3  (+4/pass)
  const int bn4 = (tid & 31) * 4;   // 0..124

  float4 aReg[8], bReg[8];

  auto load_tiles = [&](int kt) {
#pragma unroll
    for (int p = 0; p < 8; ++p) {
      aReg[p] = *(const float4*)(A + (size_t)(m0 + ar + p * 16) * K + kt + ac4);
      bReg[p] = *(const float4*)(B + (size_t)(kt + bk + p * 4) * N + n0 + bn4);
    }
  };
  auto store_tiles = [&](int buf) {
#pragma unroll
    for (int p = 0; p < 8; ++p) {
      *(bf16x4*)(&As[buf][(ar + p * 16) * 40 + ac4]) = f2bf4(aReg[p]);
      const int kk = bk + p * 4;
      Bs[buf][(bn4 + 0) * 40 + kk] = f2bf(bReg[p].x);
      Bs[buf][(bn4 + 1) * 40 + kk] = f2bf(bReg[p].y);
      Bs[buf][(bn4 + 2) * 40 + kk] = f2bf(bReg[p].z);
      Bs[buf][(bn4 + 3) * 40 + kk] = f2bf(bReg[p].w);
    }
  };

  load_tiles(0);
  store_tiles(0);
  __syncthreads();

  f32x8 acc[4][4] = {};
  const int nsteps = K >> 5;

  for (int s = 0; s < nsteps; ++s) {
    const int  cur  = s & 1;
    const bool more = (s + 1 < nsteps);
    if (more) load_tiles((s + 1) << 5);  // issue globals; waits sink past WMMAs

    bf16x16 af[4];
#pragma unroll
    for (int mf = 0; mf < 4; ++mf)
      af[mf] = frag_a(&As[cur][(wm * 64 + mf * 16 + l16) * 40], half);
#pragma unroll
    for (int nf = 0; nf < 4; ++nf) {
      bf16x16 bfv = frag_b(&Bs[cur][(wn * 64 + nf * 16 + l16) * 40], half);
#pragma unroll
      for (int mf = 0; mf < 4; ++mf)
        acc[mf][nf] = WMMA_BF16(af[mf], bfv, acc[mf][nf]);
    }
    if (more) store_tiles(cur ^ 1);
    __syncthreads();
  }

  // epilogue: C/D layout M = v + 8*half, N = lane%16
#pragma unroll
  for (int nf = 0; nf < 4; ++nf) {
    const int gcol = n0 + wn * 64 + nf * 16 + l16;
    const float badd = bias ? bias[gcol] : 0.0f;
#pragma unroll
    for (int mf = 0; mf < 4; ++mf) {
#pragma unroll
      for (int v = 0; v < 8; ++v) {
        const int grow = m0 + wm * 64 + mf * 16 + v + half * 8;
        C[(size_t)grow * N + gcol] = acc[mf][nf][v] + badd;
      }
    }
  }
}

// ---------------------------------------------------------------------------
// Flash-style causal attention: one (head, 64-query tile) per 256-thread block.
// S = (Q/8)Kᵀ and O += P·V on WMMA; K/V tiles ping-ponged in LDS with register
// prefetch of the next tile; online softmax parallelized 4 threads/row.
// 8 waves; S/O 64x64 = 16 tiles of 16x16, 2 per wave:
//   wave w -> rows (w&3)*16, cols (w>>2)*32 + {0,16}
// ---------------------------------------------------------------------------
__global__ __launch_bounds__(256) void attn_fwd_wmma(
    const float* __restrict__ qkv, float* __restrict__ attn) {
  __shared__ __align__(16) __bf16 Qs[64 * 72];      // [m][d] (pre-scaled 1/8)
  __shared__ __align__(16) __bf16 Ks[2][64 * 72];   // [k][d]
  __shared__ __align__(16) __bf16 Vts[2][64 * 72];  // [d][k] (transposed)
  __shared__ __align__(16) __bf16 Ps[64 * 72];      // [m][k] probs bf16
  __shared__ __align__(16) float  Sf[64 * 68];      // [m][k] logits f32
  __shared__ float row_max[64], row_sum[64], row_scale[64];
  __shared__ float pmax[64 * 4], psum[64 * 4];

  const int q0   = blockIdx.x * 64;
  const int h    = blockIdx.y;
  const int tid  = threadIdx.x;
  const int lane = tid & 31;
  const int wave = tid >> 5;
  const int l16  = lane & 15;
  const int half = lane >> 4;
  const int mrow = (wave & 3) * 16;
  const int ncb  = (wave >> 2) * 32;

  // staging coords (64x64 f32 tile, 4 passes of float4 per thread)
  const int sr  = tid >> 4;        // 0..15 (+16/pass)
  const int sc4 = (tid & 15) * 4;  // 0..60

  const float* kbase0 = qkv + (size_t)0 * QKV_LD + DIM + h * HD;
  const float* vbase0 = qkv + (size_t)0 * QKV_LD + 2 * DIM + h * HD;

  float4 kReg[4], vReg[4];
  auto load_kv = [&](int kt) {
#pragma unroll
    for (int p = 0; p < 4; ++p) {
      const size_t row = (size_t)(kt + sr + p * 16) * QKV_LD;
      kReg[p] = *(const float4*)(kbase0 + row + sc4);
      vReg[p] = *(const float4*)(vbase0 + row + sc4);
    }
  };
  auto store_kv = [&](int buf) {
#pragma unroll
    for (int p = 0; p < 4; ++p) {
      const int r = sr + p * 16;
      *(bf16x4*)(&Ks[buf][r * 72 + sc4]) = f2bf4(kReg[p]);
      Vts[buf][(sc4 + 0) * 72 + r] = f2bf(vReg[p].x);
      Vts[buf][(sc4 + 1) * 72 + r] = f2bf(vReg[p].y);
      Vts[buf][(sc4 + 2) * 72 + r] = f2bf(vReg[p].z);
      Vts[buf][(sc4 + 3) * 72 + r] = f2bf(vReg[p].w);
    }
  };

  // ---- prologue: stage Q (scaled), first K/V tile, init softmax state ----
  {
    const float* qbase = qkv + (size_t)q0 * QKV_LD + h * HD;
#pragma unroll
    for (int p = 0; p < 4; ++p) {
      const float4 v = *(const float4*)(qbase + (size_t)(sr + p * 16) * QKV_LD + sc4);
      bf16x4 b4 = {f2bf(v.x * 0.125f), f2bf(v.y * 0.125f),
                   f2bf(v.z * 0.125f), f2bf(v.w * 0.125f)};
      *(bf16x4*)(&Qs[(sr + p * 16) * 72 + sc4]) = b4;
    }
  }
  load_kv(0);
  store_kv(0);
  if (tid < 64) { row_max[tid] = -__builtin_inff(); row_sum[tid] = 0.0f; }
  __syncthreads();

  bf16x16 qf[2];
#pragma unroll
  for (int c = 0; c < 2; ++c)
    qf[c] = frag_a(&Qs[(mrow + l16) * 72 + c * 32], half);

  f32x8 oacc[2] = {};

  for (int kt = 0; kt <= q0; kt += 64) {
    const int  cur  = (kt >> 6) & 1;
    const bool more = (kt + 64 <= q0);
    if (more) load_kv(kt + 64);  // overlap next tile's globals with this tile

    // ---- S = Q Kᵀ ----
    f32x8 sacc[2] = {};
#pragma unroll
    for (int c = 0; c < 2; ++c) {
#pragma unroll
      for (int nf = 0; nf < 2; ++nf) {
        bf16x16 kb = frag_b(&Ks[cur][(ncb + nf * 16 + l16) * 72 + c * 32], half);
        sacc[nf] = WMMA_BF16(qf[c], kb, sacc[nf]);
      }
    }
#pragma unroll
    for (int nf = 0; nf < 2; ++nf)
#pragma unroll
      for (int v = 0; v < 8; ++v)
        Sf[(mrow + v + half * 8) * 68 + ncb + nf * 16 + l16] = sacc[nf][v];
    __syncthreads();

    // ---- online softmax, 4 threads per row ----
    const int r   = tid >> 2;
    const int seg = tid & 3;
    const int lim = (kt == q0) ? (r + 1) : 64;  // causal valid keys [0,lim)
    {
      float mx = -__builtin_inff();
#pragma unroll
      for (int j = 0; j < 16; ++j) {
        const int n = seg * 16 + j;
        if (n < lim) mx = fmaxf(mx, Sf[r * 68 + n]);
      }
      pmax[tid] = mx;
    }
    __syncthreads();
    if (tid < 64) {
      float m_old = row_max[tid];
      float m_new = fmaxf(fmaxf(fmaxf(pmax[tid * 4 + 0], pmax[tid * 4 + 1]),
                                fmaxf(pmax[tid * 4 + 2], pmax[tid * 4 + 3])),
                          m_old);
      row_scale[tid] = __expf(m_old - m_new);  // 0 on first tile (m_old=-inf)
      row_max[tid]   = m_new;
    }
    __syncthreads();
    {
      const float m_new = row_max[r];
      float s = 0.0f;
#pragma unroll
      for (int j = 0; j < 16; ++j) {
        const int n = seg * 16 + j;
        const float p = (n < lim) ? __expf(Sf[r * 68 + n] - m_new) : 0.0f;
        s += p;
        Ps[r * 72 + n] = f2bf(p);
      }
      psum[tid] = s;
    }
    __syncthreads();
    if (tid < 64)
      row_sum[tid] = row_sum[tid] * row_scale[tid] +
                     (psum[tid * 4 + 0] + psum[tid * 4 + 1]) +
                     (psum[tid * 4 + 2] + psum[tid * 4 + 3]);

    // ---- rescale O, then O += P V ----
#pragma unroll
    for (int a = 0; a < 2; ++a)
#pragma unroll
      for (int v = 0; v < 8; ++v)
        oacc[a][v] *= row_scale[mrow + v + half * 8];
#pragma unroll
    for (int c = 0; c < 2; ++c) {
      bf16x16 pf = frag_a(&Ps[(mrow + l16) * 72 + c * 32], half);
#pragma unroll
      for (int nf = 0; nf < 2; ++nf) {
        bf16x16 vb = frag_b(&Vts[cur][(ncb + nf * 16 + l16) * 72 + c * 32], half);
        oacc[nf] = WMMA_BF16(pf, vb, oacc[nf]);
      }
    }
    if (more) store_kv(cur ^ 1);
    __syncthreads();
  }

  // ---- finalize: divide by row sums, write attn[s, h*64 + d] ----
#pragma unroll
  for (int nf = 0; nf < 2; ++nf) {
    const int dcol = ncb + nf * 16 + l16;
#pragma unroll
    for (int v = 0; v < 8; ++v) {
      const int m = mrow + v + half * 8;
      attn[(size_t)(q0 + m) * DIM + h * HD + dcol] = oacc[nf][v] / row_sum[m];
    }
  }
}

// ---------------------------------------------------------------------------
// Host launcher. Workspace: qkv (2048*3072 f32) + attn (2048*1024 f32) = 32 MB.
// ---------------------------------------------------------------------------
extern "C" void kernel_launch(void* const* d_in, const int* in_sizes, int n_in,
                              void* d_out, int out_size, void* d_ws, size_t ws_size,
                              hipStream_t stream) {
  const float* x    = (const float*)d_in[0];
  const float* Wqkv = (const float*)d_in[1];
  const float* Wout = (const float*)d_in[2];
  const float* bias = (const float*)d_in[3];
  float* out  = (float*)d_out;
  float* qkv  = (float*)d_ws;                    // [2048, 3072]
  float* attn = qkv + (size_t)SEQ_LEN * QKV_LD;  // [2048, 1024]

  gemm_bf16_wmma<<<dim3(QKV_LD / 128, SEQ_LEN / 128), 128, 0, stream>>>(
      x, Wqkv, nullptr, qkv, SEQ_LEN, QKV_LD, DIM);
  attn_fwd_wmma<<<dim3(SEQ_LEN / 64, NHEAD), 256, 0, stream>>>(qkv, attn);
  gemm_bf16_wmma<<<dim3(DIM / 128, SEQ_LEN / 128), 128, 0, stream>>>(
      attn, Wout, bias, out, SEQ_LEN, DIM, DIM);
}
